// MultiHeadAttention_1511828488236
// MI455X (gfx1250) — compile-verified, matrix operand order checked by reading, not run on hardware
//
#include <hip/hip_runtime.h>
#include <hip/hip_fp16.h>
#include <stdint.h>

typedef __attribute__((ext_vector_type(16))) _Float16 v16h;
typedef __attribute__((ext_vector_type(8)))  float    v8f;

#define D_MODEL 1024
#define N_HEADS 16
#define DK      64
#define BATCH   2
#define SEQ     2048
#define NEGBIG  (-1e9f)
#define NEGINF  (-1e30f)

union AFrag { v16h h; unsigned int u[8]; };
union H16  { _Float16 h[16]; uint4 q[2]; };

// 16-bit A/B fragment K-position for element-pair v (0..7), lane-half g (0/1).
__device__ __forceinline__ int kpat(int v, int g) {
    return ((v >> 2) << 4) | (g << 3) | ((v & 3) << 1);
}

__device__ __forceinline__ v8f wmma_f16(const AFrag& a, const AFrag& b, v8f c) {
    return __builtin_amdgcn_wmma_f32_16x16x32_f16(false, a.h, false, b.h,
                                                  (short)0, c, false, false);
}

// One DPP butterfly max step (stays within each 16-lane row).
template <int CTRL>
__device__ __forceinline__ float dpp_fmax_step(float x) {
    int y = __builtin_amdgcn_mov_dpp(__builtin_bit_cast(int, x), CTRL, 0xF, 0xF, true);
    return fmaxf(x, __builtin_bit_cast(float, y));
}
// Max across the 16 lanes of each half-wave: lane^1, lane^2, lane^7, lane^15.
__device__ __forceinline__ float lane16_max(float x) {
    x = dpp_fmax_step<0xB1>(x);    // quad_perm(1,0,3,2)
    x = dpp_fmax_step<0x4E>(x);    // quad_perm(2,3,0,1)
    x = dpp_fmax_step<0x141>(x);   // row_half_mirror
    x = dpp_fmax_step<0x140>(x);   // row_mirror
    return x;
}

// Async global->LDS DMA: copy 16B per lane, tracked by ASYNCcnt.
// ldsaddr = LDS byte address (low 32 bits of generic shared pointer),
// voff    = 32-bit byte offset added to 64-bit SGPR base.
__device__ __forceinline__ void async_b128_x2(unsigned ldsaddr, unsigned voff,
                                              const void* base) {
    asm volatile("global_load_async_to_lds_b128 %0, %1, %2 offset:0"
                 :: "v"(ldsaddr), "v"(voff), "s"(base) : "memory");
    asm volatile("global_load_async_to_lds_b128 %0, %1, %2 offset:16"
                 :: "v"(ldsaddr), "v"(voff), "s"(base) : "memory");
}
__device__ __forceinline__ void async_b128_x4(unsigned ldsaddr, unsigned voff,
                                              const void* base) {
    asm volatile("global_load_async_to_lds_b128 %0, %1, %2 offset:0"
                 :: "v"(ldsaddr), "v"(voff), "s"(base) : "memory");
    asm volatile("global_load_async_to_lds_b128 %0, %1, %2 offset:16"
                 :: "v"(ldsaddr), "v"(voff), "s"(base) : "memory");
    asm volatile("global_load_async_to_lds_b128 %0, %1, %2 offset:32"
                 :: "v"(ldsaddr), "v"(voff), "s"(base) : "memory");
    asm volatile("global_load_async_to_lds_b128 %0, %1, %2 offset:48"
                 :: "v"(ldsaddr), "v"(voff), "s"(base) : "memory");
}
__device__ __forceinline__ void wait_async0() {
    asm volatile("s_wait_asynccnt 0x0" ::: "memory");
}

// ---------------------------------------------------------------------------
// Y[M,N] = X[M,K] @ W[N,K]^T + bias      M=4096, N=K=1024
// Block: 256 threads (8 waves, 4x2). Tile: 128(M) x 128(N), K-step 64.
// Each wave: 32x64 strip (2 M-subtiles x 4 N-subtiles), 16 WMMA per K-step.
// ---------------------------------------------------------------------------
template <bool IN_F16, bool OUT_F16>
__global__ __launch_bounds__(256)
void gemm_xwT_kernel(const void* __restrict__ Xv, const float* __restrict__ W,
                     const float* __restrict__ bias, void* __restrict__ Yv) {
    const int N = D_MODEL, K = D_MODEL;
    __shared__ __align__(16) _Float16 As[128][72];   // 64 + 8 pad halfs (144B rows)
    __shared__ __align__(16) _Float16 Bs[128][72];

    const int n0   = blockIdx.x * 128;
    const int m0   = blockIdx.y * 128;
    const int t    = threadIdx.x;
    const int lane = t & 31, wave = t >> 5;
    const int g    = lane >> 4, hl = lane & 15;
    const int mw   = (wave >> 1) * 32;   // 0,32,64,96
    const int nw   = (wave & 1) * 64;    // 0,64

    v8f zv = {};
    v8f acc[2][4];
#pragma unroll
    for (int s = 0; s < 2; ++s)
#pragma unroll
        for (int ns = 0; ns < 4; ++ns) acc[s][ns] = zv;

    const int srow = t >> 1;             // 0..127
    const int scol = (t & 1) * 32;       // 0 or 32

    for (int k0 = 0; k0 < K; k0 += 64) {
        // ---- stage A tile (128 x 64) -> f16 LDS ----
        if (IN_F16) {
            // Pure f16 copy: async DMA global -> LDS (4 x 16B per thread).
            const unsigned ldsA = (unsigned)(uintptr_t)&As[srow][scol];
            const unsigned voff = (unsigned)((m0 + srow) * K + k0 + scol) * 2u;
            async_b128_x4(ldsA, voff, Xv);
        } else {
            const float* src = (const float*)Xv + (size_t)(m0 + srow) * K + k0 + scol;
            float4 f[8];
#pragma unroll
            for (int i = 0; i < 8; ++i) f[i] = ((const float4*)src)[i];
            H16 pk[2];
#pragma unroll
            for (int i = 0; i < 8; ++i) {
                _Float16* hp = pk[i >> 2].h + (i & 3) * 4;
                hp[0] = (_Float16)f[i].x; hp[1] = (_Float16)f[i].y;
                hp[2] = (_Float16)f[i].z; hp[3] = (_Float16)f[i].w;
            }
            *(uint4*)&As[srow][scol]      = pk[0].q[0];
            *(uint4*)&As[srow][scol + 8]  = pk[0].q[1];
            *(uint4*)&As[srow][scol + 16] = pk[1].q[0];
            *(uint4*)&As[srow][scol + 24] = pk[1].q[1];
            if (k0 + 64 < K) __builtin_prefetch(src + 64, 0, 1);
        }
        // ---- stage B tile: W[n0+row][k0 .. k0+63] (128 x 64) ----
        {
            const float* src = W + (size_t)(n0 + srow) * K + k0 + scol;
            float4 f[8];
#pragma unroll
            for (int i = 0; i < 8; ++i) f[i] = ((const float4*)src)[i];
            H16 pk[2];
#pragma unroll
            for (int i = 0; i < 8; ++i) {
                _Float16* hp = pk[i >> 2].h + (i & 3) * 4;
                hp[0] = (_Float16)f[i].x; hp[1] = (_Float16)f[i].y;
                hp[2] = (_Float16)f[i].z; hp[3] = (_Float16)f[i].w;
            }
            *(uint4*)&Bs[srow][scol]      = pk[0].q[0];
            *(uint4*)&Bs[srow][scol + 8]  = pk[0].q[1];
            *(uint4*)&Bs[srow][scol + 16] = pk[1].q[0];
            *(uint4*)&Bs[srow][scol + 24] = pk[1].q[1];
            if (k0 + 64 < K) __builtin_prefetch(src + 64, 0, 1);
        }
        if (IN_F16) wait_async0();   // drain DMA before barrier => visible to all waves
        __syncthreads();

        // ---- compute: 2 K-chunks of 32, 8 WMMA each ----
#pragma unroll
        for (int c = 0; c < 2; ++c) {
            AFrag af[2];
#pragma unroll
            for (int s = 0; s < 2; ++s) {
                const _Float16* ap = &As[mw + s * 16 + hl][c * 32];
#pragma unroll
                for (int v = 0; v < 8; ++v)
                    af[s].u[v] = *(const unsigned int*)&ap[kpat(v, g)];
            }
#pragma unroll
            for (int ns = 0; ns < 4; ++ns) {
                AFrag bfr;
                const _Float16* bp = &Bs[nw + ns * 16 + hl][c * 32];
#pragma unroll
                for (int v = 0; v < 8; ++v)
                    bfr.u[v] = *(const unsigned int*)&bp[kpat(v, g)];
#pragma unroll
                for (int s = 0; s < 2; ++s)
                    acc[s][ns] = wmma_f16(af[s], bfr, acc[s][ns]);
            }
        }
        __syncthreads();
    }

    // ---- epilogue: + bias, store ----
#pragma unroll
    for (int ns = 0; ns < 4; ++ns) {
        const int col = n0 + nw + ns * 16 + hl;
        const float bv = bias[col];
#pragma unroll
        for (int s = 0; s < 2; ++s)
#pragma unroll
            for (int r = 0; r < 8; ++r) {
                const int row = m0 + mw + s * 16 + r + 8 * g;
                const float val = acc[s][ns][r] + bv;
                if (OUT_F16)
                    ((_Float16*)Yv)[(size_t)row * N + col] = (_Float16)val;
                else
                    ((float*)Yv)[(size_t)row * N + col] = val;
            }
    }
}

// ---------------------------------------------------------------------------
// Flash attention: softmax(Q K^T / sqrt(dk) with mask) @ V  per (b,h).
// Block: 256 threads (8 waves); each wave owns a 16-row q tile; block = 128 rows.
// Key tiles of 64. Grid: (SEQ/128, BATCH*N_HEADS).
// ---------------------------------------------------------------------------
__global__ __launch_bounds__(256)
void attention_kernel(const _Float16* __restrict__ Q, const _Float16* __restrict__ Kf,
                      const _Float16* __restrict__ Vf, const int* __restrict__ mask,
                      _Float16* __restrict__ Ctx) {
    __shared__ __align__(16) _Float16 Ks[64][72];      // [key][d], padded
    __shared__ __align__(16) _Float16 Vs[64][72];      // [d][key], padded (transposed)
    __shared__ __align__(16) _Float16 Ps[8][16][72];   // per-wave P tile (16 x 64)

    const int bh   = blockIdx.y;
    const int b    = bh >> 4, h = bh & 15;
    const int q0   = blockIdx.x * 128;
    const int t    = threadIdx.x;
    const int lane = t & 31, wave = t >> 5;
    const int g    = lane >> 4, hl = lane & 15;

    const size_t base = (size_t)b * SEQ * D_MODEL + (size_t)h * DK;
    const _Float16* Qb = Q  + base;
    const _Float16* Kb = Kf + base;
    const _Float16* Vb = Vf + base;
    const int* maskb   = mask + (size_t)b * SEQ * SEQ;

    // Load Q fragments once (two 16x32 chunks covering d_k = 64).
    AFrag qf[2];
    {
        const int qrow = q0 + wave * 16 + hl;
        const _Float16* qp = Qb + (size_t)qrow * D_MODEL;
#pragma unroll
        for (int c = 0; c < 2; ++c)
#pragma unroll
            for (int v = 0; v < 8; ++v)
                qf[c].u[v] = *(const unsigned int*)&qp[c * 32 + kpat(v, g)];
    }

    AFrag ones;   // f16 1.0 everywhere: P x Ones = row sums in C layout
#pragma unroll
    for (int v = 0; v < 8; ++v) ones.u[v] = 0x3C003C00u;

    v8f zv = {};
    v8f acc[4] = {zv, zv, zv, zv};
    float rowm[8], rowl[8];
#pragma unroll
    for (int r = 0; r < 8; ++r) { rowm[r] = NEGINF; rowl[r] = 0.f; }

    const int skey = t >> 2;           // 0..63
    const int sdc  = (t & 3) * 16;     // 0,16,32,48

    for (int kt = 0; kt < SEQ; kt += 64) {
        // ---- stage K tile [64][64] via async DMA (2 x 16B per thread) ----
        {
            const unsigned ldsK = (unsigned)(uintptr_t)&Ks[skey][sdc];
            const unsigned voff = ((unsigned)(kt + skey) * (D_MODEL * 2u)) +
                                  (unsigned)sdc * 2u;
            async_b128_x2(ldsK, voff, Kb);
        }
        // ---- stage transposed V tile [64][64] (needs VGPR transpose) ----
        {
            H16 vv;
            const uint4* vsrc = (const uint4*)(Vb + (size_t)(kt + skey) * D_MODEL + sdc);
            vv.q[0] = vsrc[0];
            vv.q[1] = vsrc[1];
#pragma unroll
            for (int j = 0; j < 16; ++j) Vs[sdc + j][skey] = vv.h[j];
        }
        wait_async0();                 // K DMA done before barrier
        __syncthreads();

        // ---- scores: 16(q) x 64(key), k-dim = d_k in two chunks ----
        v8f s[4];
#pragma unroll
        for (int ks = 0; ks < 4; ++ks) {
            s[ks] = zv;
#pragma unroll
            for (int c = 0; c < 2; ++c) {
                AFrag bk;
                const _Float16* kp = &Ks[ks * 16 + hl][c * 32];
#pragma unroll
                for (int v = 0; v < 8; ++v)
                    bk.u[v] = *(const unsigned int*)&kp[kpat(v, g)];
                s[ks] = wmma_f16(qf[c], bk, s[ks]);
            }
        }

        // ---- scale + mask ----
#pragma unroll
        for (int ks = 0; ks < 4; ++ks) {
            const int key = kt + ks * 16 + hl;
#pragma unroll
            for (int r = 0; r < 8; ++r) {
                const int qr = q0 + wave * 16 + r + 8 * g;
                const int mv = maskb[(size_t)qr * SEQ + key];
                s[ks][r] = (mv == 0) ? NEGBIG : s[ks][r] * 0.125f;
            }
        }

        // ---- row max via DPP butterfly (no DS ops, no waits) ----
        float alpha[8];
#pragma unroll
        for (int r = 0; r < 8; ++r) {
            float tm = fmaxf(fmaxf(s[0][r], s[1][r]), fmaxf(s[2][r], s[3][r]));
            tm = lane16_max(tm);
            const float mnew = fmaxf(rowm[r], tm);
            alpha[r] = (rowm[r] == NEGINF) ? 0.f : __expf(rowm[r] - mnew);
            rowm[r] = mnew;
        }
#pragma unroll
        for (int ks = 0; ks < 4; ++ks)
#pragma unroll
            for (int r = 0; r < 8; ++r)
                s[ks][r] = __expf(s[ks][r] - rowm[r]);

        // ---- relayout P (C layout) -> A fragments via wave-private LDS ----
#pragma unroll
        for (int ks = 0; ks < 4; ++ks)
#pragma unroll
            for (int r = 0; r < 8; ++r)
                Ps[wave][r + 8 * g][ks * 16 + hl] = (_Float16)s[ks][r];
        __asm__ volatile("s_wait_dscnt 0" ::: "memory");

        AFrag pf[2];
#pragma unroll
        for (int c = 0; c < 2; ++c) {
            const _Float16* pp = &Ps[wave][hl][c * 32];
#pragma unroll
            for (int v = 0; v < 8; ++v)
                pf[c].u[v] = *(const unsigned int*)&pp[kpat(v, g)];
        }

        // ---- row sums via WMMA with ones matrix (replaces shuffle-sum) ----
        v8f lsum = wmma_f16(pf[0], ones, zv);
        lsum = wmma_f16(pf[1], ones, lsum);
#pragma unroll
        for (int r = 0; r < 8; ++r) rowl[r] = rowl[r] * alpha[r] + lsum[r];

        // ---- rescale accumulators, then acc += P @ V ----
#pragma unroll
        for (int ns = 0; ns < 4; ++ns)
#pragma unroll
            for (int r = 0; r < 8; ++r) acc[ns][r] *= alpha[r];
#pragma unroll
        for (int ns = 0; ns < 4; ++ns) {
#pragma unroll
            for (int c = 0; c < 2; ++c) {
                AFrag vfr;
                const _Float16* vp = &Vs[ns * 16 + hl][c * 32];
#pragma unroll
                for (int v = 0; v < 8; ++v)
                    vfr.u[v] = *(const unsigned int*)&vp[kpat(v, g)];
                acc[ns] = wmma_f16(pf[c], vfr, acc[ns]);
            }
        }
        __syncthreads();
    }

    // ---- normalize and store context (f16, [b][s][h*64+d]) ----
    float inv[8];
#pragma unroll
    for (int r = 0; r < 8; ++r) inv[r] = (rowl[r] > 0.f) ? (1.f / rowl[r]) : 0.f;
#pragma unroll
    for (int ns = 0; ns < 4; ++ns) {
        const int d = ns * 16 + hl;
#pragma unroll
        for (int r = 0; r < 8; ++r) {
            const int qr = q0 + wave * 16 + r + 8 * g;
            Ctx[base + (size_t)qr * D_MODEL + d] = (_Float16)(acc[ns][r] * inv[r]);
        }
    }
}

// ---------------------------------------------------------------------------
extern "C" void kernel_launch(void* const* d_in, const int* in_sizes, int n_in,
                              void* d_out, int out_size, void* d_ws, size_t ws_size,
                              hipStream_t stream) {
    const float* query = (const float*)d_in[0];
    const float* key_i = (const float*)d_in[1];
    const float* value = (const float*)d_in[2];
    const int*   mask  = (const int*)d_in[3];
    const float* W_q = (const float*)d_in[4];  const float* b_q = (const float*)d_in[5];
    const float* W_k = (const float*)d_in[6];  const float* b_k = (const float*)d_in[7];
    const float* W_v = (const float*)d_in[8];  const float* b_v = (const float*)d_in[9];
    const float* W_o = (const float*)d_in[10]; const float* b_o = (const float*)d_in[11];
    float* out = (float*)d_out;

    const size_t tsz = (size_t)BATCH * SEQ * D_MODEL;   // elements per tensor
    _Float16* Qh = (_Float16*)d_ws;
    _Float16* Kh = Qh + tsz;
    _Float16* Vh = Kh + tsz;
    _Float16* Ch = Vh + tsz;

    dim3 gg(D_MODEL / 128, (BATCH * SEQ) / 128);   // 8 x 32
    gemm_xwT_kernel<false, true><<<gg, 256, 0, stream>>>(query, W_q, b_q, Qh);
    gemm_xwT_kernel<false, true><<<gg, 256, 0, stream>>>(key_i, W_k, b_k, Kh);
    gemm_xwT_kernel<false, true><<<gg, 256, 0, stream>>>(value, W_v, b_v, Vh);

    dim3 ga(SEQ / 128, BATCH * N_HEADS);           // 16 x 32
    attention_kernel<<<ga, 256, 0, stream>>>(Qh, Kh, Vh, mask, Ch);

    gemm_xwT_kernel<true, false><<<gg, 256, 0, stream>>>(Ch, W_o, b_o, out);
}
